// GroupedQueryAttention_35931696398604
// MI455X (gfx1250) — matrix operand order repaired and checked by reading, not compiled, not run on hardware
//
#include <hip/hip_runtime.h>
#include <hip/hip_bf16.h>

// ---------------------------------------------------------------------------
// Types for CDNA5 WMMA (gfx1250, wave32)
// ---------------------------------------------------------------------------
typedef __attribute__((ext_vector_type(16))) __bf16 v16bf;
typedef __attribute__((ext_vector_type(8)))  float  v8f;

// Matches the builtin prototype pointee: gcc-style vector of 4 ints (b128).
typedef int gqa_i4 __attribute__((vector_size(16)));

union Frag16 {
    v16bf v;
    uint4 q[2];
    unsigned short u[16];
};

__device__ __forceinline__ unsigned short f32_to_bf16(float f) {
    unsigned int u = __builtin_bit_cast(unsigned int, f);
    u += 0x7FFFu + ((u >> 16) & 1u);     // round-to-nearest-even
    return (unsigned short)(u >> 16);
}

// A-matrix (16x32 bf16) striping per ISA 7.12.2:
//   e<8 : k = 8*g + e      e>=8 : k = 16 + 8*g + (e-8)      (g = lane>>4)
__device__ __forceinline__ v16bf load_frag_a(const unsigned short* row_k0, int g) {
    Frag16 f;
    f.q[0] = *reinterpret_cast<const uint4*>(row_k0 + 8 * g);
    f.q[1] = *reinterpret_cast<const uint4*>(row_k0 + 16 + 8 * g);
    return f.v;
}

// B-matrix (32x16 bf16) striping, same k order as A:
//   e<8 : k = 8*g + e      e>=8 : k = 16 + 8*g + (e-8)
__device__ __forceinline__ v16bf load_frag_b(const unsigned short* col_k0, int g) {
    Frag16 f;
    f.q[0] = *reinterpret_cast<const uint4*>(col_k0 + 8 * g);
    f.q[1] = *reinterpret_cast<const uint4*>(col_k0 + 16 + 8 * g);
    return f.v;
}

// ---------------------------------------------------------------------------
// Async global->LDS copy (ASYNCcnt-tracked) with sync fallback
// ---------------------------------------------------------------------------
#if defined(__has_builtin)
#if __has_builtin(__builtin_amdgcn_global_load_async_to_lds_b128) && \
    __has_builtin(__builtin_amdgcn_s_wait_asynccnt)
#define GQA_ASYNC_LDS 1
#endif
#endif
#ifndef GQA_ASYNC_LDS
#define GQA_ASYNC_LDS 0
#endif

__device__ __forceinline__ void copy16_g2l(const unsigned short* g, unsigned short* l) {
#if GQA_ASYNC_LDS
    __builtin_amdgcn_global_load_async_to_lds_b128((gqa_i4*)g, (gqa_i4*)l, 0, 0);
#else
    *reinterpret_cast<uint4*>(l) = *reinterpret_cast<const uint4*>(g);
#endif
}

__device__ __forceinline__ void tile_arrive_barrier() {
#if GQA_ASYNC_LDS
    __builtin_amdgcn_s_wait_asynccnt(0);   // my async copies landed in LDS
#endif
    __syncthreads();                        // everyone's copies landed
}

// ---------------------------------------------------------------------------
// Problem constants
// ---------------------------------------------------------------------------
#define BATCH 2
#define SEQ 2048
#define DMODEL 2048
#define NHEADS 32
#define NKV 8
#define HDIM 64
#define KVDIM (NKV * HDIM)            // 512
#define QKVC (DMODEL + 2 * KVDIM)     // 3072

// ---------------------------------------------------------------------------
// f32 -> bf16 conversion (plain, and transposing for weights)
// ---------------------------------------------------------------------------
__global__ void cvt_bf16_kernel(const float* __restrict__ src,
                                unsigned short* __restrict__ dst, int n) {
    int i = blockIdx.x * blockDim.x + threadIdx.x;
    if (i < n) dst[i] = f32_to_bf16(src[i]);
}

// src [K,N] f32 row-major -> dst [N,K] bf16 row-major
__global__ void cvt_bf16_tr_kernel(const float* __restrict__ src,
                                   unsigned short* __restrict__ dst, int K, int N) {
    int i = blockIdx.x * blockDim.x + threadIdx.x;
    if (i >= K * N) return;
    int k = i / N, n = i - k * N;
    dst[(size_t)n * K + k] = f32_to_bf16(src[i]);
}

// ---------------------------------------------------------------------------
// Tiled WMMA GEMM: C[M,N] f32 = A[M,K] * Bt[N,K]^T   (both bf16, k-major)
// Block tile 64x128, BK=32, 256 threads (8 waves, 2x4), 32x32 per wave.
// Double-buffered LDS, async global->LDS staging.
// ---------------------------------------------------------------------------
#define BM 64
#define BN 128
#define BK 32

template <int M, int N, int K>
__global__ __launch_bounds__(256)
void gemm_bf16_f32_kernel(const unsigned short* __restrict__ A,
                          const unsigned short* __restrict__ Bt,
                          float* __restrict__ C) {
    __shared__ __align__(16) unsigned short As[2][BM][BK];   // [m][k]
    __shared__ __align__(16) unsigned short Bs[2][BN][BK];   // [n][k]

    const int t    = threadIdx.x;
    const int lane = t & 31;
    const int w    = t >> 5;
    const int g    = lane >> 4;
    const int l16  = lane & 15;
    const int wm   = (w >> 2) * 32;          // 0 / 32
    const int wn   = (w & 3) * 32;           // 0..96
    const int m0   = blockIdx.y * BM;
    const int n0   = blockIdx.x * BN;

    // A loader: 8 elems/thread (1 x b128); B loader: 16 elems/thread (2 x b128)
    const int ar  = t >> 2;                  // 0..63
    const int ak  = (t & 3) * 8;             // 0..24
    const int br  = t >> 1;                  // 0..127
    const int bkc = (t & 1) * 16;            // 0 / 16

    const unsigned short* Ag = &A[(size_t)(m0 + ar) * K + ak];
    const unsigned short* Bg = &Bt[(size_t)(n0 + br) * K + bkc];

    v8f acc00 = {}, acc01 = {}, acc10 = {}, acc11 = {};

    // prologue: stage tile 0 into buffer 0
    copy16_g2l(Ag, &As[0][ar][ak]);
    copy16_g2l(Bg, &Bs[0][br][bkc]);
    copy16_g2l(Bg + 8, &Bs[0][br][bkc + 8]);

    int cur = 0;
    for (int k0 = 0; k0 < K; k0 += BK) {
        tile_arrive_barrier();               // tile(k0) resident in LDS[cur]

        if (k0 + BK < K) {                   // stage next tile into other buffer
            int nxt = cur ^ 1;
            copy16_g2l(Ag + k0 + BK, &As[nxt][ar][ak]);
            copy16_g2l(Bg + k0 + BK, &Bs[nxt][br][bkc]);
            copy16_g2l(Bg + k0 + BK + 8, &Bs[nxt][br][bkc + 8]);
#if !GQA_ASYNC_LDS
            __builtin_prefetch(Ag + k0 + 2 * BK, 0, 3);
            __builtin_prefetch(Bg + k0 + 2 * BK, 0, 3);
#endif
        }

        v16bf a0 = load_frag_a(&As[cur][wm + l16][0], g);
        v16bf a1 = load_frag_a(&As[cur][wm + 16 + l16][0], g);
        v16bf b0 = load_frag_b(&Bs[cur][wn + l16][0], g);
        v16bf b1 = load_frag_b(&Bs[cur][wn + 16 + l16][0], g);

        acc00 = __builtin_amdgcn_wmma_f32_16x16x32_bf16(false, a0, false, b0, (short)0, acc00, false, false);
        acc01 = __builtin_amdgcn_wmma_f32_16x16x32_bf16(false, a0, false, b1, (short)0, acc01, false, false);
        acc10 = __builtin_amdgcn_wmma_f32_16x16x32_bf16(false, a1, false, b0, (short)0, acc10, false, false);
        acc11 = __builtin_amdgcn_wmma_f32_16x16x32_bf16(false, a1, false, b1, (short)0, acc11, false, false);
        cur ^= 1;
    }

    // C/D layout: VGPR r -> row (r + 8g), col = l16 within each 16x16 tile.
    float* c0 = C + (size_t)(m0 + wm + 8 * g) * N + (n0 + wn + l16);
#pragma unroll
    for (int r = 0; r < 8; ++r) {
        c0[(size_t)r * N]             = acc00[r];
        c0[(size_t)r * N + 16]        = acc01[r];
        c0[(size_t)(r + 16) * N]      = acc10[r];
        c0[(size_t)(r + 16) * N + 16] = acc11[r];
    }
}

// ---------------------------------------------------------------------------
// RoPE + split + layout kernel.
// qkv f32 [B,T,3072] -> q bf16 [B,H,T,64], k bf16 [B,KV,T,64],
//                       vT bf16 [B,KV,64,T]
// ---------------------------------------------------------------------------
__global__ void rope_split_kernel(const float* __restrict__ qkv,
                                  const float* __restrict__ sinp,
                                  const float* __restrict__ cosp,
                                  unsigned short* __restrict__ qb,
                                  unsigned short* __restrict__ kb,
                                  unsigned short* __restrict__ vtb) {
    int idx = blockIdx.x * blockDim.x + threadIdx.x;
    const int total = BATCH * SEQ * QKVC;
    if (idx >= total) return;
    int c  = idx % QKVC;
    int bt = idx / QKVC;
    int t  = bt % SEQ;
    int b  = bt / SEQ;
    const float* row = qkv + (size_t)bt * QKVC;

    if (c < DMODEL) {                         // Q + RoPE
        int h = c >> 6, d = c & 63;
        float u   = row[c];
        float rot = (d < 32) ? -row[c + 32] : row[c - 32];
        float o   = u * cosp[t * HDIM + d] + rot * sinp[t * HDIM + d];
        qb[((size_t)(b * NHEADS + h) * SEQ + t) * HDIM + d] = f32_to_bf16(o);
    } else if (c < DMODEL + KVDIM) {          // K + RoPE
        int c2 = c - DMODEL;
        int kv = c2 >> 6, d = c2 & 63;
        float u   = row[c];
        float rot = (d < 32) ? -row[c + 32] : row[c - 32];
        float o   = u * cosp[t * HDIM + d] + rot * sinp[t * HDIM + d];
        kb[((size_t)(b * NKV + kv) * SEQ + t) * HDIM + d] = f32_to_bf16(o);
    } else {                                  // V, stored transposed [d][t]
        int c2 = c - DMODEL - KVDIM;
        int kv = c2 >> 6, d = c2 & 63;
        vtb[((size_t)(b * NKV + kv) * HDIM + d) * SEQ + t] = f32_to_bf16(row[c]);
    }
}

// ---------------------------------------------------------------------------
// Flash-attention (causal, GQA).  One wave per 16-query tile; 32-key blocks.
// 8 WMMAs per key block (4 for QK^T, 4 for PV).
// ---------------------------------------------------------------------------
__global__ __launch_bounds__(256)
void attn_kernel(const unsigned short* __restrict__ Q,    // [B,H,T,64]
                 const unsigned short* __restrict__ Km,   // [B,KV,T,64]
                 const unsigned short* __restrict__ Vt,   // [B,KV,64,T]
                 unsigned short* __restrict__ O) {        // [B,T,H*64]
    __shared__ __align__(16) unsigned short pbuf[8][16 * 32];

    const int w    = threadIdx.x >> 5;
    const int lane = threadIdx.x & 31;
    const int g    = lane >> 4;
    const int l16  = lane & 15;

    const int grp = blockIdx.x;                 // 0 .. B*H*16-1
    const int b   = grp / (NHEADS * 16);
    const int h   = (grp / 16) % NHEADS;
    const int tg  = grp % 16;
    const int q0  = (tg * 8 + w) * 16;          // query tile start
    const int kvh = h >> 2;                     // H/KV = 4

    const unsigned short* Qb = Q  + ((size_t)(b * NHEADS + h) * SEQ + q0) * HDIM;
    const unsigned short* Kb = Km + (size_t)(b * NKV + kvh) * SEQ * HDIM;
    const unsigned short* Vb = Vt + (size_t)(b * NKV + kvh) * HDIM * SEQ;

    v16bf aq0 = load_frag_a(Qb + l16 * HDIM, g);        // d = 0..31
    v16bf aq1 = load_frag_a(Qb + l16 * HDIM + 32, g);   // d = 32..63

    v8f accO0 = {}, accO1 = {}, accO2 = {}, accO3 = {};
    float m[8], l[8];
#pragma unroll
    for (int r = 0; r < 8; ++r) { m[r] = -3.0e38f; l[r] = 0.0f; }
    const float scale = 0.125f;                 // 64^-0.5
    unsigned short* Pw = pbuf[w];

    for (int kb = 0; kb <= q0 + 15; kb += 32) {
        // ---- S = Q K^T  (two 16x16 score tiles over 32 keys) ----
        v8f s0 = {}, s1 = {};
#pragma unroll
        for (int ka = 0; ka < 2; ++ka) {
            v16bf aq  = ka ? aq1 : aq0;
            v16bf bk0 = load_frag_b(Kb + (size_t)(kb + l16) * HDIM + 32 * ka, g);
            v16bf bk1 = load_frag_b(Kb + (size_t)(kb + 16 + l16) * HDIM + 32 * ka, g);
            s0 = __builtin_amdgcn_wmma_f32_16x16x32_bf16(false, aq, false, bk0, (short)0, s0, false, false);
            s1 = __builtin_amdgcn_wmma_f32_16x16x32_bf16(false, aq, false, bk1, (short)0, s1, false, false);
        }

        // ---- online softmax over the 32-key block ----
        const int col0 = kb + l16;
        const int col1 = kb + 16 + l16;
#pragma unroll
        for (int r = 0; r < 8; ++r) {
            int   qrow = q0 + r + 8 * g;
            float a  = (col0 <= qrow) ? s0[r] * scale : -3.0e38f;
            float bb = (col1 <= qrow) ? s1[r] * scale : -3.0e38f;
            float mx = fmaxf(a, bb);
            mx = fmaxf(mx, __shfl_xor(mx, 1));
            mx = fmaxf(mx, __shfl_xor(mx, 2));
            mx = fmaxf(mx, __shfl_xor(mx, 4));
            mx = fmaxf(mx, __shfl_xor(mx, 8));
            float mn    = fmaxf(m[r], mx);
            float alpha = __expf(m[r] - mn);
            float p0    = __expf(a - mn);
            float p1    = __expf(bb - mn);
            float rs    = p0 + p1;
            rs += __shfl_xor(rs, 1);
            rs += __shfl_xor(rs, 2);
            rs += __shfl_xor(rs, 4);
            rs += __shfl_xor(rs, 8);
            l[r] = l[r] * alpha + rs;
            m[r] = mn;
            // restripe P (C-layout) into LDS row-major [16 q][32 k]
            int prow = r + 8 * g;
            Pw[prow * 32 + l16]      = f32_to_bf16(p0);
            Pw[prow * 32 + 16 + l16] = f32_to_bf16(p1);
            accO0[r] *= alpha; accO1[r] *= alpha; accO2[r] *= alpha; accO3[r] *= alpha;
        }

        // ---- O += P V  (A fragment re-read from LDS in A-striping) ----
        v16bf ap = load_frag_a(&Pw[l16 * 32], g);
        v16bf bv0 = load_frag_b(Vb + (size_t)(0  + l16) * SEQ + kb, g);
        v16bf bv1 = load_frag_b(Vb + (size_t)(16 + l16) * SEQ + kb, g);
        v16bf bv2 = load_frag_b(Vb + (size_t)(32 + l16) * SEQ + kb, g);
        v16bf bv3 = load_frag_b(Vb + (size_t)(48 + l16) * SEQ + kb, g);
        accO0 = __builtin_amdgcn_wmma_f32_16x16x32_bf16(false, ap, false, bv0, (short)0, accO0, false, false);
        accO1 = __builtin_amdgcn_wmma_f32_16x16x32_bf16(false, ap, false, bv1, (short)0, accO1, false, false);
        accO2 = __builtin_amdgcn_wmma_f32_16x16x32_bf16(false, ap, false, bv2, (short)0, accO2, false, false);
        accO3 = __builtin_amdgcn_wmma_f32_16x16x32_bf16(false, ap, false, bv3, (short)0, accO3, false, false);
    }

    // ---- finalize: O /= l, write bf16 [B,T,H*64] ----
#pragma unroll
    for (int r = 0; r < 8; ++r) {
        float inv  = 1.0f / l[r];
        int   trow = q0 + r + 8 * g;
        size_t base = ((size_t)b * SEQ + trow) * DMODEL + h * HDIM;
        O[base + 0  + l16] = f32_to_bf16(accO0[r] * inv);
        O[base + 16 + l16] = f32_to_bf16(accO1[r] * inv);
        O[base + 32 + l16] = f32_to_bf16(accO2[r] * inv);
        O[base + 48 + l16] = f32_to_bf16(accO3[r] * inv);
    }
}

// ---------------------------------------------------------------------------
// Host-side orchestration
// ---------------------------------------------------------------------------
extern "C" void kernel_launch(void* const* d_in, const int* in_sizes, int n_in,
                              void* d_out, int out_size, void* d_ws, size_t ws_size,
                              hipStream_t stream) {
    (void)in_sizes; (void)n_in; (void)out_size; (void)ws_size;

    const float* x     = (const float*)d_in[0];
    const float* sinp  = (const float*)d_in[1];
    const float* cosp  = (const float*)d_in[2];
    const float* w_qkv = (const float*)d_in[3];
    const float* w_out = (const float*)d_in[4];
    float* out = (float*)d_out;

    char*  ws  = (char*)d_ws;
    size_t off = 0;
    auto alloc = [&](size_t bytes) -> void* {
        void* p = ws + off;
        off = (off + bytes + 255) & ~(size_t)255;
        return p;
    };

    const size_t nX    = (size_t)BATCH * SEQ * DMODEL;           // 8388608
    const size_t nWqkv = (size_t)DMODEL * QKVC;                  // 6291456
    const size_t nWout = (size_t)DMODEL * DMODEL;                // 4194304
    const size_t nQKV  = (size_t)BATCH * SEQ * QKVC;             // 12582912

    unsigned short* xb    = (unsigned short*)alloc(nX * 2);
    unsigned short* wqkvt = (unsigned short*)alloc(nWqkv * 2);   // [QKVC, D]
    unsigned short* woutt = (unsigned short*)alloc(nWout * 2);   // [D, D] transposed
    float*          qkvf  = (float*)alloc(nQKV * 4);
    unsigned short* qb    = (unsigned short*)alloc((size_t)BATCH * NHEADS * SEQ * HDIM * 2);
    unsigned short* kbuf  = (unsigned short*)alloc((size_t)BATCH * NKV * SEQ * HDIM * 2);
    unsigned short* vtb   = (unsigned short*)alloc((size_t)BATCH * NKV * HDIM * SEQ * 2);
    unsigned short* ob    = (unsigned short*)alloc(nX * 2);

    // 1) convert inputs to bf16 (weights transposed to k-major for the GEMMs)
    cvt_bf16_kernel<<<(int)((nX + 255) / 256), 256, 0, stream>>>(x, xb, (int)nX);
    cvt_bf16_tr_kernel<<<(int)((nWqkv + 255) / 256), 256, 0, stream>>>(w_qkv, wqkvt, DMODEL, QKVC);
    cvt_bf16_tr_kernel<<<(int)((nWout + 255) / 256), 256, 0, stream>>>(w_out, woutt, DMODEL, DMODEL);

    // 2) QKV projection: [4096,2048] x [2048,3072] -> f32
    {
        dim3 grid(QKVC / BN, (BATCH * SEQ) / BM);
        gemm_bf16_f32_kernel<BATCH * SEQ, QKVC, DMODEL>
            <<<grid, 256, 0, stream>>>(xb, wqkvt, qkvf);
    }

    // 3) RoPE + split + V-transpose
    rope_split_kernel<<<(int)((nQKV + 255) / 256), 256, 0, stream>>>(
        qkvf, sinp, cosp, qb, kbuf, vtb);

    // 4) causal GQA flash attention -> bf16 [B,T,D]
    attn_kernel<<<BATCH * NHEADS * 16, 256, 0, stream>>>(qb, kbuf, vtb, ob);

    // 5) output projection: [4096,2048] x [2048,2048] -> f32 d_out
    {
        dim3 grid(DMODEL / BN, (BATCH * SEQ) / BM);
        gemm_bf16_f32_kernel<BATCH * SEQ, DMODEL, DMODEL>
            <<<grid, 256, 0, stream>>>(ob, woutt, out);
    }
}
